// GraphAttentionLayer_3135326126645
// MI455X (gfx1250) — matrix operand order repaired and checked by reading, not compiled, not run on hardware
//
#include <hip/hip_runtime.h>
#include <hip/hip_bf16.h>

// ---------------------------------------------------------------------------
// GAT layer, fused for gfx1250 (MI455X): wave32, bf16 WMMA (f32 accumulate),
// Tensor Data Mover double-buffered LDS staging for the PV GEMM B-panel.
// N=4096, F_IN=256, H=4, D=64.
// ---------------------------------------------------------------------------

typedef __attribute__((ext_vector_type(16))) __bf16       v16bf;
typedef __attribute__((ext_vector_type(8)))  float        v8f;
typedef __attribute__((ext_vector_type(8)))  unsigned int v8u;
typedef __attribute__((ext_vector_type(4)))  unsigned int v4u;
typedef __attribute__((ext_vector_type(8)))  int          v8i;
typedef __attribute__((ext_vector_type(4)))  int          v4i;

#define GN   4096
#define GFIN 256
#define GH   4
#define GD   64
#define GALPHA 0.2f

#define TJ    128            // j-columns staged per TDM chunk
#define NCH   (GN / TJ)      // 32 chunks
#define PITCH 272            // LDS bytes per d-row: 256 data + 16 TDM pad

__device__ __forceinline__ unsigned short f2bf(float x) {
  union { float f; unsigned int u; } v; v.f = x;
  unsigned int u = v.u;
  unsigned int r = u + 0x7fffu + ((u >> 16) & 1u);   // round-nearest-even
  return (unsigned short)(r >> 16);
}

__device__ __forceinline__ v8f wmma_bf16(v8u a, v8u b, v8f c) {
  v16bf av = __builtin_bit_cast(v16bf, a);
  v16bf bv = __builtin_bit_cast(v16bf, b);
  return __builtin_amdgcn_wmma_f32_16x16x32_bf16(false, av, false, bv,
                                                 (short)0, c, false, false);
}

// TDM: stage a (GD=64 rows) x (TJ=128 cols) bf16 panel of Wh^T into LDS.
// D# per CDNA5 ISA:
//   group0: count=1, lds_addr, global_addr[56:0], type=2
//   group1: data_size=2B, pad_enable (interval 64 DW = one 256B row,
//           amount 4 DW = 16B -> LDS pitch 272B), tensor 4096x64,
//           tile 128x64, dim0_stride 4096 elements.
// This toolchain's builtin takes 6 args (extra int32x8 group, then cpol).
__device__ __forceinline__ void tdm_load_panel(const unsigned short* gbase,
                                               unsigned ldsOff) {
  unsigned long long ga = (unsigned long long)(uintptr_t)gbase;
  v4u g0;
  g0[0] = 1u;                                            // count=1, user D#
  g0[1] = ldsOff;                                        // lds_addr
  g0[2] = (unsigned)ga;                                  // global_addr[31:0]
  g0[3] = ((unsigned)(ga >> 32) & 0x01FFFFFFu)           // global_addr[56:32]
          | (2u << 30);                                  // type=2 (image)
  v8i g1;
  g1[0] = (int)((1u << 16)        // data_size = 1 -> 2 bytes
              | (1u << 20)        // pad_enable
              | (5u << 22)        // pad_interval: 64 DWORDs (one tile row)
              | (3u << 25));      // pad_amount: 4 DWORDs = 16 bytes
  g1[1] = (int)(((unsigned)GN & 0xFFFFu) << 16);         // tensor_dim0 lo16
  g1[2] = (int)(((unsigned)GN >> 16) | ((unsigned)GD << 16)); // dim0 hi / dim1
  g1[3] = (int)((unsigned)TJ << 16);                     // tile_dim0 = 128
  g1[4] = (int)GD;                                       // tile_dim1 = 64
  g1[5] = (int)GN;                                       // dim0_stride lo32
  g1[6] = 0;
  g1[7] = 0;
  v4i g2 = {};                                           // 2D: groups 2/3 zero
  v4i g3 = {};
  v8i g4 = {};                                           // extra group (zero)
  __builtin_amdgcn_tensor_load_to_lds(g0, g1, g2, g3, g4, 0);
}

// --------------------------- stage 0: conversions ---------------------------

__global__ void __launch_bounds__(256) gat_cvt_h(const float* __restrict__ h,
                                                 unsigned short* __restrict__ hB,
                                                 int n) {
  int idx = blockIdx.x * 256 + threadIdx.x;
  if (idx < n) hB[idx] = f2bf(h[idx]);
}

// W: (H, F_IN, D) f32  ->  WT: (H, D, F_IN) bf16 (transposed for WMMA-B reads)
__global__ void __launch_bounds__(256) gat_cvt_w(const float* __restrict__ W,
                                                 unsigned short* __restrict__ WT) {
  int idx = blockIdx.x * 256 + threadIdx.x;           // H*F_IN*D = 65536
  int head = idx >> 14;
  int rem  = idx & 16383;
  int k = rem >> 6;
  int d = rem & 63;
  WT[(head * GD + d) * GFIN + k] = f2bf(W[idx]);
}

// ------------------- stage 1: Wh GEMM + per-node scores ---------------------

__global__ void __launch_bounds__(256) gat_wh_gemm(
    const unsigned short* __restrict__ hB,    // (N, F_IN) bf16
    const unsigned short* __restrict__ WT,    // (H, D, F_IN) bf16
    const float* __restrict__ a_src,          // (H, D)
    const float* __restrict__ a_dst,          // (H, D)
    unsigned short* __restrict__ WhT,         // (H, D, N) bf16
    float* __restrict__ s_src,                // (H, N)
    float* __restrict__ s_dst) {              // (H, N)
  __shared__ float lds[8][16 * 68];           // per-wave 16x64 tile, pad 68
  const int wave = threadIdx.x >> 5;
  const int lane = threadIdx.x & 31;
  const int gw   = blockIdx.x * 8 + wave;     // 0..1023
  const int head = gw >> 8;
  const int i0   = (gw & 255) * 16;
  const int rA   = lane & 15;
  const int hiH  = lane >> 4;

  v8f c[4] = {};

  for (int k0 = 0; k0 < GFIN; k0 += 32) {
    v8u a;
#pragma unroll
    for (int v = 0; v < 8; ++v) {             // A: 16x32 bf16 (contiguous k = 16h + s)
      int kk = hiH * 16 + v * 2;
      a[v] = *(const unsigned int*)(hB + (size_t)(i0 + rA) * GFIN + k0 + kk);
    }
#pragma unroll
    for (int b = 0; b < 4; ++b) {             // B: 32x16 bf16 per d-block
      v8u bb;
#pragma unroll
      for (int v = 0; v < 8; ++v) {
        bb[v] = *(const unsigned int*)(
            WT + (size_t)(head * GD + b * 16 + rA) * GFIN + k0 + hiH * 16 + 2 * v);
      }
      c[b] = wmma_bf16(a, bb, c[b]);
    }
  }

  // Stage C tile (C/D layout: N = lane%16, M = 8*laneGroup + vgpr) into LDS.
  float* T = &lds[wave][0];
  const int mrow = hiH * 8;
#pragma unroll
  for (int b = 0; b < 4; ++b)
#pragma unroll
    for (int v = 0; v < 8; ++v)
      T[(mrow + v) * 68 + b * 16 + rA] = c[b][v];
  __syncthreads();

  // Per-row scores: half-wave splits d-range, one xor-shuffle combines.
  float ss = 0.f, sd = 0.f;
  const float* asrc = a_src + head * GD;
  const float* adst = a_dst + head * GD;
  for (int d = hiH * 32; d < hiH * 32 + 32; ++d) {
    float w = T[rA * 68 + d];
    ss += w * asrc[d];
    sd += w * adst[d];
  }
  ss += __shfl_xor(ss, 16, 32);
  sd += __shfl_xor(sd, 16, 32);
  if (hiH == 0) {
    s_src[head * GN + i0 + rA] = ss;
    s_dst[head * GN + i0 + rA] = sd;
  }

  // Store Wh^T (H, D, N) bf16 for the PV GEMM's B operand.
  for (int t = lane; t < 16 * GD; t += 32) {
    int d = t >> 4, r = t & 15;
    WhT[(size_t)(head * GD + d) * GN + i0 + r] = f2bf(T[r * 68 + d]);
  }
}

// ------------- stage 2: masked softmax stats (one adj sweep) ----------------

__global__ void __launch_bounds__(256) gat_stats(
    const float* __restrict__ adj,
    const float* __restrict__ s_src, const float* __restrict__ s_dst,
    float* __restrict__ m_out, float* __restrict__ l_out) {
  const int wave = threadIdx.x >> 5;
  const int lane = threadIdx.x & 31;
  const int i = blockIdx.x * 8 + wave;
  const float* adjRow = adj + (size_t)i * GN;

  float src[GH], m[GH], l[GH];
#pragma unroll
  for (int h = 0; h < GH; ++h) {
    src[h] = s_src[h * GN + i];
    m[h] = -9e15f;                             // matches reference mask fill
    l[h] = 0.f;
  }

  for (int j = lane; j < GN; j += 32) {
    __builtin_prefetch(adjRow + j + 2048, 0, 1);   // global_prefetch_b8 ahead
    float a = adjRow[j];
    if (a > 0.f) {
#pragma unroll
      for (int h = 0; h < GH; ++h) {
        float e = src[h] + s_dst[h * GN + j];
        e = (e > 0.f) ? e : GALPHA * e;        // leaky_relu
        if (e > m[h]) {
          l[h] = l[h] * __expf(m[h] - e) + 1.f;
          m[h] = e;
        } else {
          l[h] += __expf(e - m[h]);
        }
      }
    }
  }
#pragma unroll
  for (int off = 16; off >= 1; off >>= 1) {
#pragma unroll
    for (int h = 0; h < GH; ++h) {
      float mo = __shfl_xor(m[h], off, 32);
      float lo = __shfl_xor(l[h], off, 32);
      float mn = fmaxf(m[h], mo);
      l[h] = l[h] * __expf(m[h] - mn) + lo * __expf(mo - mn);
      m[h] = mn;
    }
  }
  if (lane == 0) {
#pragma unroll
    for (int h = 0; h < GH; ++h) {
      m_out[h * GN + i] = m[h];
      l_out[h * GN + i] = l[h];
    }
  }
}

// -------------- stage 3: fused P·Wh WMMA GEMM + ELU epilogue ----------------
// 8 waves/block share one head; the Wh^T B-panel is staged into LDS by the
// Tensor Data Mover (double-buffered on TENSORcnt) so the VMEM pipe carries
// only the streaming adj reads. Each wave owns a 16(i) x 64(d) output tile.

__global__ void __launch_bounds__(256) gat_attn_pv(
    const float* __restrict__ adj,
    const unsigned short* __restrict__ WhT,   // (H, D, N) bf16
    const float* __restrict__ s_src, const float* __restrict__ s_dst,
    const float* __restrict__ m_in, const float* __restrict__ l_in,
    float* __restrict__ out) {                // (N, H*D)
  __shared__ unsigned char bbuf[2][GD * PITCH];  // 2 x 17408 B staged panels
  __shared__ float rowstat[8][3][16];            // per-wave {src, m, 1/l}
  const int wave = threadIdx.x >> 5;
  const int lane = threadIdx.x & 31;
  const int gw   = blockIdx.x * 8 + wave;
  const int head = gw >> 8;                      // same for all 8 waves
  const int i0   = (gw & 255) * 16;
  const int rA   = lane & 15;
  const int hiH  = lane >> 4;

  if (lane < 16) {
    int i = i0 + lane;
    rowstat[wave][0][lane] = s_src[head * GN + i];
    rowstat[wave][1][lane] = m_in[head * GN + i];
    float l = l_in[head * GN + i];
    rowstat[wave][2][lane] = (l > 0.f) ? 1.0f / l : 0.0f;
  }

  const float* adjRow = adj + (size_t)(i0 + rA) * GN;
  const float* dstH   = s_dst + head * GN;
  const unsigned short* WB = WhT + (size_t)head * GD * GN;
  const unsigned ldsB0 = (unsigned)(uintptr_t)&bbuf[0][0];
  const unsigned ldsB1 = (unsigned)(uintptr_t)&bbuf[1][0];

  if (wave == 0) tdm_load_panel(WB, ldsB0);      // prime chunk 0
  __syncthreads();                               // rowstat visible too

  const float srcR = rowstat[wave][0][rA];
  const float mR   = rowstat[wave][1][rA];

  v8f c[4] = {};

  for (int ch = 0; ch < NCH; ++ch) {
    if (wave == 0) {
      if (ch + 1 < NCH) {                        // overlap next DMA w/ compute
        tdm_load_panel(WB + (ch + 1) * TJ, (ch & 1) ? ldsB0 : ldsB1);
        __builtin_amdgcn_s_wait_tensorcnt(1);    // chunk ch landed
      } else {
        __builtin_amdgcn_s_wait_tensorcnt(0);
      }
    }
    __syncthreads();                             // panel ready for all waves
    const unsigned char* buf = (ch & 1) ? &bbuf[1][0] : &bbuf[0][0];

#pragma unroll
    for (int s = 0; s < 4; ++s) {                // 4 x 32-j subtiles
      const int jt = ch * TJ + s * 32;
      v8u A;
#pragma unroll
      for (int v = 0; v < 8; ++v) {              // P pairs in WMMA-A layout
        int kk = hiH * 16 + v * 2;
        int j  = jt + kk;
        float2 av = *(const float2*)(adjRow + j);
        float2 dv = *(const float2*)(dstH + j);
        float e0 = srcR + dv.x; e0 = (e0 > 0.f) ? e0 : GALPHA * e0;
        float e1 = srcR + dv.y; e1 = (e1 > 0.f) ? e1 : GALPHA * e1;
        float p0 = (av.x > 0.f) ? __expf(e0 - mR) : 0.f;
        float p1 = (av.y > 0.f) ? __expf(e1 - mR) : 0.f;
        A[v] = (unsigned int)f2bf(p0) | ((unsigned int)f2bf(p1) << 16);
      }
#pragma unroll
      for (int b = 0; b < 4; ++b) {              // B from LDS: 32 contig bytes
        const unsigned char* p = buf + (b * 16 + rA) * PITCH + s * 64 + hiH * 32;
        uint4 lo = *(const uint4*)(p);
        uint4 hi = *(const uint4*)(p + 16);
        v8u B;
        B[0] = lo.x; B[1] = lo.y; B[2] = lo.z; B[3] = lo.w;
        B[4] = hi.x; B[5] = hi.y; B[6] = hi.z; B[7] = hi.w;
        c[b] = wmma_bf16(A, B, c[b]);
      }
    }
    __syncthreads();                             // fence before buffer reuse
  }

  // epilogue: scale by 1/l, ELU, scatter to (N, H*D) output
  const int mrow = hiH * 8;
#pragma unroll
  for (int b = 0; b < 4; ++b)
#pragma unroll
    for (int v = 0; v < 8; ++v) {
      int r = mrow + v;
      float val = c[b][v] * rowstat[wave][2][r];
      val = (val > 0.f) ? val : (__expf(val) - 1.f);   // ELU(alpha=1)
      out[(size_t)(i0 + r) * (GH * GD) + head * GD + b * 16 + rA] = val;
    }
}

// ---------------------------------------------------------------------------

extern "C" void kernel_launch(void* const* d_in, const int* in_sizes, int n_in,
                              void* d_out, int out_size, void* d_ws, size_t ws_size,
                              hipStream_t stream) {
  (void)in_sizes; (void)n_in; (void)out_size; (void)ws_size;
  const float* h     = (const float*)d_in[0];   // (4096, 256)
  const float* adj   = (const float*)d_in[1];   // (4096, 4096)
  const float* W     = (const float*)d_in[2];   // (4, 256, 64)
  const float* a_src = (const float*)d_in[3];   // (4, 64, 1)
  const float* a_dst = (const float*)d_in[4];   // (4, 64, 1)
  float* out = (float*)d_out;                   // (4096, 256)

  char* ws = (char*)d_ws;
  unsigned short* hB  = (unsigned short*)ws; ws += (size_t)GN * GFIN * 2;
  unsigned short* WT  = (unsigned short*)ws; ws += (size_t)GH * GD * GFIN * 2;
  unsigned short* WhT = (unsigned short*)ws; ws += (size_t)GH * GD * GN * 2;
  float* s_src = (float*)ws; ws += (size_t)GH * GN * 4;
  float* s_dst = (float*)ws; ws += (size_t)GH * GN * 4;
  float* m_buf = (float*)ws; ws += (size_t)GH * GN * 4;
  float* l_buf = (float*)ws; ws += (size_t)GH * GN * 4;

  gat_cvt_h<<<(GN * GFIN + 255) / 256, 256, 0, stream>>>(h, hB, GN * GFIN);
  gat_cvt_w<<<(GH * GFIN * GD) / 256, 256, 0, stream>>>(W, WT);
  gat_wh_gemm<<<128, 256, 0, stream>>>(hB, WT, a_src, a_dst, WhT, s_src, s_dst);
  gat_stats<<<GN / 8, 256, 0, stream>>>(adj, s_src, s_dst, m_buf, l_buf);
  gat_attn_pv<<<128, 256, 0, stream>>>(adj, WhT, s_src, s_dst, m_buf, l_buf, out);
}
